// PanoContrastiveLoss_56358560858353
// MI455X (gfx1250) — matrix operand order repaired and checked by reading, not compiled, not run on hardware
//
#include <hip/hip_runtime.h>
#include <hip/hip_bf16.h>
#include <stdint.h>

// ---------------------------------------------------------------------------
// PanoContrastiveLoss on MI455X (gfx1250, wave32, WMMA bf16)
//
//   L = sym_infonce(Q Kt / 0.07)  +  0.1 * sym_infonce(z1 z2t / 0.2)
//   sym_infonce = 0.5*mean(lse rows) + 0.5*mean(lse cols) - mean(diag)
//
// max logit <= 1/0.07 ~= 14.3 -> exp < 1.6e6 -> single-pass sum(exp) is safe.
// One wave computes a 64x64 logits block as 4x4 WMMA tiles (16 v_wmma per
// K-step, 8 fragment loads -> 0.5 frag-loads/wmma). f32->bf16 conversion is
// a plain fptrunc so the backend can select packed HW converts. Cross-lane
// reductions via ds_swizzle XOR butterflies, f32 global atomics into row/col
// exp-sum accumulators, then a tiny log+mean finish kernel.
// ---------------------------------------------------------------------------

typedef __bf16 v16bf __attribute__((ext_vector_type(16)));
typedef float  v8f   __attribute__((ext_vector_type(8)));

#define BB 8
#define VV 8
#define SS 576
#define DD 768
#define HH 24
#define WW 24
#define KO 12                   // overlap width k = W/2
#define NOV (BB*HH*KO)          // 2304 rows per overlap matrix
#define BLK64_OV (NOV/64)       // 36 wave-blocks per dim
#define BLK64_TL (SS/64)        // 9  wave-blocks per dim
#define INV_TAU_OV (1.0f/0.07f)
#define INV_TAU_TL (1.0f/0.2f)

// workspace layout (float elements)
#define WS_OV_DIAG 0
#define WS_TL_DIAG 1
#define WS_OV_ROW  2
#define WS_OV_COL  (WS_OV_ROW + VV*NOV)          // +18432
#define WS_TL_ROW  (WS_OV_COL + VV*NOV)          // +18432
#define WS_TL_COL  (WS_TL_ROW + BB*VV*SS)        // +36864
#define WS_TOTAL   (WS_TL_COL + BB*VV*SS)        // 110594 floats (~443 KB)

// Plain fptrunc: backend selects packed bf16 HW converts when present,
// else emits the standard RNE software sequence. No target-gated builtins.
static __device__ __forceinline__ __bf16 f2bf(float f) { return (__bf16)f; }

template <int E0>
static __device__ __forceinline__ void load8(const float* __restrict__ p, v16bf& d) {
  float4 x = *reinterpret_cast<const float4*>(p);
  float4 y = *reinterpret_cast<const float4*>(p + 4);
  d[E0 + 0] = f2bf(x.x); d[E0 + 1] = f2bf(x.y);
  d[E0 + 2] = f2bf(x.z); d[E0 + 3] = f2bf(x.w);
  d[E0 + 4] = f2bf(y.x); d[E0 + 5] = f2bf(y.y);
  d[E0 + 6] = f2bf(y.z); d[E0 + 7] = f2bf(y.w);
}

// ds_swizzle group-of-32 XOR butterfly: offset = {xor[4:0], or[4:0]=0, and[4:0]=0x1f}
template <int IMM>
static __device__ __forceinline__ float xor_add(float v) {
  int x = __builtin_amdgcn_ds_swizzle(__builtin_bit_cast(int, v), IMM);
  return v + __builtin_bit_cast(float, x);
}
#define SWZ_X1  0x041F
#define SWZ_X2  0x081F
#define SWZ_X4  0x101F
#define SWZ_X8  0x201F
#define SWZ_X16 0x401F

// One wave: 64x64 logits block = 4x4 tiles of v_wmma_f32_16x16x32_bf16.
// qrow[i]/krow[i] are per-lane row pointers: A row = rowBase+i*16+(lane&15).
static __device__ __forceinline__ void tile_core4(
    const float* const (&qrow)[4], const float* const (&krow)[4],
    float invTau,
    float* __restrict__ rowsum, float* __restrict__ colsum,
    int rowBase, int colBase, bool isDiag, float* __restrict__ diagAcc)
{
  const int lane = (int)(threadIdx.x & 31u);
  const int hv   = lane >> 4;
  const int col  = lane & 15;

  v8f acc[4][4];
#pragma unroll
  for (int i = 0; i < 4; ++i)
#pragma unroll
    for (int j = 0; j < 4; ++j)
      acc[i][j] = (v8f){0.f, 0.f, 0.f, 0.f, 0.f, 0.f, 0.f, 0.f};

#pragma unroll 1
  for (int k0 = 0; k0 < DD; k0 += 32) {
    v16bf a[4], b[4];
#pragma unroll
    for (int i = 0; i < 4; ++i) {
      // A 16x32 bf16: lanes 0-15 hold K {0..7,16..23}, lanes 16-31 {8..15,24..31}
      load8<0>(qrow[i] + k0 + 8 * hv,      a[i]);
      load8<8>(qrow[i] + k0 + 16 + 8 * hv, a[i]);
    }
#pragma unroll
    for (int j = 0; j < 4; ++j) {
      // B 32x16 bf16: lanes 0-15 hold K {0..7,16..23}, lanes 16-31 {8..15,24..31}
      load8<0>(krow[j] + k0 + 8 * hv,      b[j]);
      load8<8>(krow[j] + k0 + 16 + 8 * hv, b[j]);
    }
#pragma unroll
    for (int i = 0; i < 4; ++i)
#pragma unroll
      for (int j = 0; j < 4; ++j)
        acc[i][j] = __builtin_amdgcn_wmma_f32_16x16x32_bf16(
                        false, a[i], false, b[j], (short)0, acc[i][j],
                        false, false);
  }

  // C layout: element (lane, r) of tile (i,j) -> row = i*16 + r + 8*hv,
  //                                              col = j*16 + (lane&15)
  float rs[4][8];
  float cs[4] = {0.f, 0.f, 0.f, 0.f};
  float dsum  = 0.f;
#pragma unroll
  for (int i = 0; i < 4; ++i)
#pragma unroll
    for (int r = 0; r < 8; ++r)
      rs[i][r] = 0.f;

#pragma unroll
  for (int i = 0; i < 4; ++i)
#pragma unroll
    for (int j = 0; j < 4; ++j)
#pragma unroll
      for (int r = 0; r < 8; ++r) {
        float l = acc[i][j][r] * invTau;
        float e = __expf(l);
        rs[i][r] += e;
        cs[j]    += e;
        if (isDiag && i == j && (r + 8 * hv) == col) dsum += l;
      }

  // row exp-sums: butterfly across the 16 lanes of each half
#pragma unroll
  for (int i = 0; i < 4; ++i)
#pragma unroll
    for (int r = 0; r < 8; ++r) {
      float s = rs[i][r];
      s = xor_add<SWZ_X1>(s);
      s = xor_add<SWZ_X2>(s);
      s = xor_add<SWZ_X4>(s);
      s = xor_add<SWZ_X8>(s);
      if (col == 0) atomicAdd(&rowsum[rowBase + i * 16 + r + 8 * hv], s);
    }

  // column exp-sums: fold the two halves
#pragma unroll
  for (int j = 0; j < 4; ++j) {
    float s = xor_add<SWZ_X16>(cs[j]);
    if (hv == 0) atomicAdd(&colsum[colBase + j * 16 + col], s);
  }

  // diagonal logits: full-wave reduce, one atomic
  if (isDiag) {
    float d = dsum;
    d = xor_add<SWZ_X1>(d);
    d = xor_add<SWZ_X2>(d);
    d = xor_add<SWZ_X4>(d);
    d = xor_add<SWZ_X8>(d);
    d = xor_add<SWZ_X16>(d);
    if (lane == 0) atomicAdd(diagAcc, d);
  }
}

// ---------------- overlap loss: V matrices of 2304x2304 -----------------
__global__ void __launch_bounds__(256)
ov_kernel(const float* __restrict__ z1, const float* __restrict__ z2,
          float* __restrict__ ws)
{
  const int waveId = (int)blockIdx.x * 8 + (int)(threadIdx.x >> 5);
  const int tn = waveId % BLK64_OV;
  int t        = waveId / BLK64_OV;
  const int tm = t % BLK64_OV;
  const int v  = t / BLK64_OV;
  const int lr = (int)(threadIdx.x & 15u);

  const float* qrow[4];
  const float* krow[4];
#pragma unroll
  for (int i = 0; i < 4; ++i) {
    // Q row m: right half of view v:  z1[b*V+v, h*W + (W-KO) + x, :]
    int m  = tm * 64 + i * 16 + lr;
    int b1 = m / (HH * KO);
    int r1 = m % (HH * KO);
    int h1 = r1 / KO;
    int x1 = r1 % KO;
    qrow[i] = z1 + (size_t)((b1 * VV + v) * SS + h1 * WW + (WW - KO) + x1) * DD;
    // K row n: left half of rolled view (v+1)%V:  z2[b*V+(v+1)%V, h*W + x, :]
    int n  = tn * 64 + i * 16 + lr;
    int b2 = n / (HH * KO);
    int r2 = n % (HH * KO);
    int h2 = r2 / KO;
    int x2 = r2 % KO;
    krow[i] = z2 + (size_t)((b2 * VV + ((v + 1) & (VV - 1))) * SS + h2 * WW + x2) * DD;
  }

  tile_core4(qrow, krow, INV_TAU_OV,
             ws + WS_OV_ROW + (size_t)v * NOV,
             ws + WS_OV_COL + (size_t)v * NOV,
             tm * 64, tn * 64, tm == tn, ws + WS_OV_DIAG);
}

// ---------------- tile loss: 64 matrices of 576x576 ----------------------
__global__ void __launch_bounds__(256)
tl_kernel(const float* __restrict__ z1, const float* __restrict__ z2,
          float* __restrict__ ws)
{
  const int waveId = (int)blockIdx.x * 8 + (int)(threadIdx.x >> 5);
  const int tn = waveId % BLK64_TL;
  int t        = waveId / BLK64_TL;
  const int tm = t % BLK64_TL;
  const int bv = t / (BLK64_TL);
  const int lr = (int)(threadIdx.x & 15u);

  const float* qrow[4];
  const float* krow[4];
#pragma unroll
  for (int i = 0; i < 4; ++i) {
    qrow[i] = z1 + (size_t)(bv * SS + tm * 64 + i * 16 + lr) * DD;
    krow[i] = z2 + (size_t)(bv * SS + tn * 64 + i * 16 + lr) * DD;
  }

  tile_core4(qrow, krow, INV_TAU_TL,
             ws + WS_TL_ROW + (size_t)bv * SS,
             ws + WS_TL_COL + (size_t)bv * SS,
             tm * 64, tn * 64, tm == tn, ws + WS_TL_DIAG);
}

// ---------------- final reduction: logs, means, combine ------------------
__global__ void __launch_bounds__(256)
finish_kernel(const float* __restrict__ ws, float* __restrict__ out)
{
  const int tid = (int)threadIdx.x;
  float s_ovr = 0.f, s_ovc = 0.f, s_tlr = 0.f, s_tlc = 0.f;
  for (int i = tid; i < VV * NOV; i += 256) {
    s_ovr += __logf(ws[WS_OV_ROW + i]);
    s_ovc += __logf(ws[WS_OV_COL + i]);
  }
  for (int i = tid; i < BB * VV * SS; i += 256) {
    s_tlr += __logf(ws[WS_TL_ROW + i]);
    s_tlc += __logf(ws[WS_TL_COL + i]);
  }
  __shared__ float red[4][256];
  red[0][tid] = s_ovr; red[1][tid] = s_ovc;
  red[2][tid] = s_tlr; red[3][tid] = s_tlc;
  __syncthreads();
  for (int off = 128; off > 0; off >>= 1) {
    if (tid < off) {
      red[0][tid] += red[0][tid + off];
      red[1][tid] += red[1][tid + off];
      red[2][tid] += red[2][tid + off];
      red[3][tid] += red[3][tid + off];
    }
    __syncthreads();
  }
  if (tid == 0) {
    const float nov = (float)(VV * NOV);       // 18432
    const float ntl = (float)(BB * VV * SS);   // 36864
    float L_ov = 0.5f * (red[0][0] + red[1][0]) / nov - ws[WS_OV_DIAG] / nov;
    float L_tl = 0.5f * (red[2][0] + red[3][0]) / ntl - ws[WS_TL_DIAG] / ntl;
    out[0] = L_ov + 0.1f * L_tl;
  }
}

extern "C" void kernel_launch(void* const* d_in, const int* in_sizes, int n_in,
                              void* d_out, int out_size, void* d_ws, size_t ws_size,
                              hipStream_t stream)
{
  (void)in_sizes; (void)n_in; (void)out_size; (void)ws_size;
  const float* z1 = (const float*)d_in[0];
  const float* z2 = (const float*)d_in[1];
  float* ws  = (float*)d_ws;
  float* out = (float*)d_out;

  hipMemsetAsync(d_ws, 0, (size_t)WS_TOTAL * sizeof(float), stream);

  // overlap: 8 * 36 * 36 = 10368 waves, 8 waves (256 thr) per block
  const int ovBlocks = (VV * BLK64_OV * BLK64_OV) / 8;       // 1296
  ov_kernel<<<ovBlocks, 256, 0, stream>>>(z1, z2, ws);

  // tile: 64 * 9 * 9 = 5184 waves
  const int tlBlocks = (BB * VV * BLK64_TL * BLK64_TL) / 8;  // 648
  tl_kernel<<<tlBlocks, 256, 0, stream>>>(z1, z2, ws);

  finish_kernel<<<1, 256, 0, stream>>>(ws, out);
}
